// ArcMarginProduct_28355374088215
// MI455X (gfx1250) — matrix operand order repaired and checked by reading, not compiled, not run on hardware
//
#include <hip/hip_runtime.h>
#include <hip/hip_bf16.h>

typedef _Float16 v4h  __attribute__((ext_vector_type(4)));
typedef _Float16 v8h  __attribute__((ext_vector_type(8)));
typedef _Float16 v16h __attribute__((ext_vector_type(16)));
typedef float    v8f  __attribute__((ext_vector_type(8)));

#define N_ROWS 512
#define DIM    512
#define NCLS   100000

#define BN 32               // classes per block
#define KSTEPS (DIM / 32)   // 16

// ArcFace constants: s=30, m=0.5
#define ARC_S     30.0f
#define ARC_COS_M 0.8775825618903728f
#define ARC_SIN_M 0.47942553860420306f
#define ARC_TH    (-0.8775825618903728f)   /* cos(pi - m) */
#define ARC_MM    0.23971276930210154f     /* sin(pi - m) * m */

// ---------------------------------------------------------------------------
// Prepass: L2-normalize each input row (fp32) and store as f16 into workspace.
// One block (256 threads) per row of 512 elements.
// ---------------------------------------------------------------------------
__global__ void __launch_bounds__(256)
normalize_input_f16(const float* __restrict__ x, _Float16* __restrict__ xn) {
  __shared__ float red[256];
  __shared__ float sInv;
  const int row = blockIdx.x;
  const int t   = threadIdx.x;

  const float a = x[(size_t)row * DIM + t];
  const float b = x[(size_t)row * DIM + 256 + t];
  red[t] = a * a + b * b;
  __syncthreads();
  for (int s = 128; s > 0; s >>= 1) {
    if (t < s) red[t] += red[t + s];
    __syncthreads();
  }
  if (t == 0) sInv = 1.0f / fmaxf(sqrtf(red[0]), 1e-12f);
  __syncthreads();
  const float inv = sInv;
  xn[(size_t)row * DIM + t]       = (_Float16)(a * inv);
  xn[(size_t)row * DIM + 256 + t] = (_Float16)(b * inv);
}

// ---------------------------------------------------------------------------
// Fused GEMM + weight-norm + ArcFace, fragments loaded directly from global.
// Block: 256 threads = 8 waves; block tile = all 512 rows x 32 classes.
// Each wave: 4 M-subtiles x 2 N-subtiles = 64x32 outputs, 8 accumulators.
// No LDS staging, no K-loop barriers: A fragments are two b128 loads/lane
// from L2-resident Xn; B fragments are 16 fp32 weight values/lane converted
// to f16 in registers (weight row touched by exactly one block -> HBM once).
// Per-class sum(w^2) accumulated per lane, folded with one shfl_xor(16).
// ---------------------------------------------------------------------------
__global__ void __launch_bounds__(256)
arc_margin_gemm(const _Float16* __restrict__ xn,
                const float* __restrict__ w,
                const long long* __restrict__ label,
                float* __restrict__ out) {
  __shared__ int sLab[N_ROWS];

  const int t    = threadIdx.x;
  const int lane = t & 31;
  const int wave = t >> 5;
  const int blockCol = blockIdx.x * BN;

  sLab[t]       = (int)label[t];
  sLab[t + 256] = (int)label[t + 256];
  __syncthreads();

  const int mlane = lane & 15;
  const int hi    = lane >> 4;          // lane half
  const int mbase = wave * 64;          // this wave's 64-row group
  const int ak0   = hi ? 16 : 0;        // A: K 0-15 / 16-31
  const int bk0   = hi ? 16 : 0;        // B: K 0-15 / 16-31

  // A fragment sources: row = mbase + i*16 + mlane (f16, K contiguous).
  const _Float16* __restrict__ ap0 = xn + (size_t)(mbase +  0 + mlane) * DIM;
  const _Float16* __restrict__ ap1 = xn + (size_t)(mbase + 16 + mlane) * DIM;
  const _Float16* __restrict__ ap2 = xn + (size_t)(mbase + 32 + mlane) * DIM;
  const _Float16* __restrict__ ap3 = xn + (size_t)(mbase + 48 + mlane) * DIM;

  // B fragment sources: class = blockCol + j*16 + mlane (fp32, K contiguous).
  const float* __restrict__ bp0 = w + (size_t)(blockCol +  0 + mlane) * DIM;
  const float* __restrict__ bp1 = w + (size_t)(blockCol + 16 + mlane) * DIM;

  float ssq0 = 0.0f, ssq1 = 0.0f;
  v8f a00 = {}, a01 = {}, a10 = {}, a11 = {},
      a20 = {}, a21 = {}, a30 = {}, a31 = {};

#pragma unroll 2
  for (int kt = 0; kt < KSTEPS; ++kt) {
    const int ka = kt * 32 + ak0;
    const int kb = kt * 32 + bk0;

    // ---- B fragments: 16 fp32 -> 16 f16 per lane, accumulate sum(w^2) ----
    v16h bv0, bv1;
    {
      const float4 f0 = *(const float4*)(bp0 + kb);
      const float4 f1 = *(const float4*)(bp0 + kb + 4);
      const float4 f2 = *(const float4*)(bp0 + kb + 8);
      const float4 f3 = *(const float4*)(bp0 + kb + 12);
      ssq0 += f0.x*f0.x + f0.y*f0.y + f0.z*f0.z + f0.w*f0.w
            + f1.x*f1.x + f1.y*f1.y + f1.z*f1.z + f1.w*f1.w
            + f2.x*f2.x + f2.y*f2.y + f2.z*f2.z + f2.w*f2.w
            + f3.x*f3.x + f3.y*f3.y + f3.z*f3.z + f3.w*f3.w;
      bv0[0]=(_Float16)f0.x;  bv0[1]=(_Float16)f0.y;
      bv0[2]=(_Float16)f0.z;  bv0[3]=(_Float16)f0.w;
      bv0[4]=(_Float16)f1.x;  bv0[5]=(_Float16)f1.y;
      bv0[6]=(_Float16)f1.z;  bv0[7]=(_Float16)f1.w;
      bv0[8]=(_Float16)f2.x;  bv0[9]=(_Float16)f2.y;
      bv0[10]=(_Float16)f2.z; bv0[11]=(_Float16)f2.w;
      bv0[12]=(_Float16)f3.x; bv0[13]=(_Float16)f3.y;
      bv0[14]=(_Float16)f3.z; bv0[15]=(_Float16)f3.w;
    }
    {
      const float4 f0 = *(const float4*)(bp1 + kb);
      const float4 f1 = *(const float4*)(bp1 + kb + 4);
      const float4 f2 = *(const float4*)(bp1 + kb + 8);
      const float4 f3 = *(const float4*)(bp1 + kb + 12);
      ssq1 += f0.x*f0.x + f0.y*f0.y + f0.z*f0.z + f0.w*f0.w
            + f1.x*f1.x + f1.y*f1.y + f1.z*f1.z + f1.w*f1.w
            + f2.x*f2.x + f2.y*f2.y + f2.z*f2.z + f2.w*f2.w
            + f3.x*f3.x + f3.y*f3.y + f3.z*f3.z + f3.w*f3.w;
      bv1[0]=(_Float16)f0.x;  bv1[1]=(_Float16)f0.y;
      bv1[2]=(_Float16)f0.z;  bv1[3]=(_Float16)f0.w;
      bv1[4]=(_Float16)f1.x;  bv1[5]=(_Float16)f1.y;
      bv1[6]=(_Float16)f1.z;  bv1[7]=(_Float16)f1.w;
      bv1[8]=(_Float16)f2.x;  bv1[9]=(_Float16)f2.y;
      bv1[10]=(_Float16)f2.z; bv1[11]=(_Float16)f2.w;
      bv1[12]=(_Float16)f3.x; bv1[13]=(_Float16)f3.y;
      bv1[14]=(_Float16)f3.z; bv1[15]=(_Float16)f3.w;
    }

    // ---- A fragments (direct b128 loads) + 8 WMMAs ----
#define DO_ROW(AP, ACCJ0, ACCJ1)                                              \
    {                                                                         \
      const v8h alo = *(const v8h*)((AP) + ka);                               \
      const v8h ahi = *(const v8h*)((AP) + ka + 8);                           \
      const v16h av = __builtin_shufflevector(alo, ahi,                       \
          0, 1, 2, 3, 4, 5, 6, 7, 8, 9, 10, 11, 12, 13, 14, 15);              \
      ACCJ0 = __builtin_amdgcn_wmma_f32_16x16x32_f16(                         \
          false, av, false, bv0, (short)0, ACCJ0, false, false);              \
      ACCJ1 = __builtin_amdgcn_wmma_f32_16x16x32_f16(                         \
          false, av, false, bv1, (short)0, ACCJ1, false, false);              \
    }
    DO_ROW(ap0, a00, a01)
    DO_ROW(ap1, a10, a11)
    DO_ROW(ap2, a20, a21)
    DO_ROW(ap3, a30, a31)
#undef DO_ROW
  }

  // Fold the two K-halves of sum(w^2) across lane halves.
  ssq0 += __shfl_xor(ssq0, 16, 32);
  ssq1 += __shfl_xor(ssq1, 16, 32);
  const float winv0 = 1.0f / fmaxf(sqrtf(ssq0), 1e-12f);
  const float winv1 = 1.0f / fmaxf(sqrtf(ssq1), 1e-12f);

  // Epilogue: ArcFace margin, label select, scale; non-temporal stores.
  const int rsub = hi << 3;   // C/D layout: VGPR r -> row r / r+8 by lane half
  const int col0 = blockCol + mlane;
  const int col1 = blockCol + 16 + mlane;
  v8f accs[4][2] = {{a00, a01}, {a10, a11}, {a20, a21}, {a30, a31}};
#pragma unroll
  for (int i = 0; i < 4; ++i) {
    const int mtile = mbase + i * 16 + rsub;
#pragma unroll
    for (int r = 0; r < 8; ++r) {
      const int lrow = mtile + r;
      const int lab  = sLab[lrow];
      {
        const float cosv = accs[i][0][r] * winv0;
        const float sine = sqrtf(fmaxf(1.0f - cosv * cosv, 0.0f));
        float phi = cosv * ARC_COS_M - sine * ARC_SIN_M;
        phi = (cosv > ARC_TH) ? phi : (cosv - ARC_MM);
        const float val = ((lab == col0) ? phi : cosv) * ARC_S;
        __builtin_nontemporal_store(val, &out[(size_t)lrow * NCLS + col0]);
      }
      {
        const float cosv = accs[i][1][r] * winv1;
        const float sine = sqrtf(fmaxf(1.0f - cosv * cosv, 0.0f));
        float phi = cosv * ARC_COS_M - sine * ARC_SIN_M;
        phi = (cosv > ARC_TH) ? phi : (cosv - ARC_MM);
        const float val = ((lab == col1) ? phi : cosv) * ARC_S;
        __builtin_nontemporal_store(val, &out[(size_t)lrow * NCLS + col1]);
      }
    }
  }
}

// ---------------------------------------------------------------------------
extern "C" void kernel_launch(void* const* d_in, const int* in_sizes, int n_in,
                              void* d_out, int out_size, void* d_ws, size_t ws_size,
                              hipStream_t stream) {
  const float*     x   = (const float*)d_in[0];      // [512, 512] fp32
  const long long* lab = (const long long*)d_in[1];  // [512] int64
  const float*     w   = (const float*)d_in[2];      // [100000, 512] fp32
  float*           out = (float*)d_out;              // [512, 100000] fp32

  _Float16* xn = (_Float16*)d_ws;  // 512*512*2 = 512 KB of workspace

  normalize_input_f16<<<N_ROWS, 256, 0, stream>>>(x, xn);

  // One block per 32-class tile; each block covers all 512 rows.
  arc_margin_gemm<<<NCLS / BN, 256, 0, stream>>>(xn, w, lab, out);
}
